// DynEdge_19335942766754
// MI455X (gfx1250) — compile-verified, matrix operand order checked by reading, not run on hardware
//
#include <hip/hip_runtime.h>
#include <stdint.h>

// ---------------------------------------------------------------------------
// DynamicEdgeConv (DGCNN) forward for MI455X / gfx1250, f16 WMMA matrix path.
// B=64 graphs x N=512 points, K=8 neighbors, 4 edge-conv layers + point MLP.
// Compute-bound (~450 GFLOP vs ~100 MB traffic) -> v_wmma_f32_16x16x32_f16.
// Round 2: B-fragment reuse across both M-tiles (halves L2 weight traffic),
// vectorized LDS staging of gathered rows (kills scalar u16 global loads),
// stage buffer aliased with the hidden-activation buffer.
// ---------------------------------------------------------------------------

typedef __attribute__((ext_vector_type(16))) _Float16 v16h;
typedef __attribute__((ext_vector_type(8)))  _Float16 v8h;
typedef __attribute__((ext_vector_type(8)))  float    v8f;

#define NBATCH 64
#define NPT    512
#define NPTS   (NBATCH * NPT)   // 32768 points
#define KNB    8                // neighbors per point
#define DH     336              // hidden width
#define DOUT   256              // per-layer output width
#define NT1    22               // 352/16 n-tiles, GEMM1
#define KT2    11               // 352/32 k-tiles, GEMM2
#define NT2    16               // 256/16 n-tiles, GEMM2

static __device__ __forceinline__ float lrelu(float x) {
  return x >= 0.f ? x : 0.01f * x;
}

static __device__ __forceinline__ v8f wmma16(v16h a, v16h b, v8f c) {
  // D = A(16x32 f16) * B(32x16 f16) + C(16x16 f32)
  return __builtin_amdgcn_wmma_f32_16x16x32_f16(false, a, false, b,
                                                (short)0, c, false, false);
}

// Fragment-order packing for 16-bit A/B tiles (ISA 7.12.2, 16-bit A 16x32):
//   lane = (row&15) | (((k>>3)&1)<<4) ;  half = ((k>>4)<<3) | (k&7)
// Inverse (used by packers):  k = (h&7) | ((lane>>4)<<3) | ((h>>3)<<4)

// ---- pack an f32 [din x dout] weight matrix into fragment-ordered f16 -----
__global__ __launch_bounds__(256) void pack_w_kernel(
    const float* __restrict__ src, _Float16* __restrict__ dst,
    int din, int dout, int kt_cnt, int total) {
  int e = blockIdx.x * 256 + threadIdx.x;
  if (e >= total) return;
  int h  = e & 15;
  int ln = (e >> 4) & 31;
  int kt = (e >> 9) % kt_cnt;
  int nt = (e >> 9) / kt_cnt;
  int n  = nt * 16 + (ln & 15);
  int k  = (h & 7) | ((ln >> 4) << 3) | ((h >> 3) << 4);
  int kg = kt * 32 + k;
  float v = (kg < din && n < dout) ? src[(size_t)kg * dout + n] : 0.f;
  dst[e] = (_Float16)v;
}

// ---- convert x [32768 x 6] f32 -> [32768 x 16] f16 (zero padded) ----------
__global__ __launch_bounds__(256) void cvt_x0_kernel(
    const float* __restrict__ x, _Float16* __restrict__ xp) {
  int e = blockIdx.x * 256 + threadIdx.x;        // over 32768*16
  int i = e >> 4, c = e & 15;
  xp[e] = (c < 6) ? (_Float16)x[i * 6 + c] : (_Float16)0.f;
}

// ---- kNN (top-8 smallest squared distance, self included) -----------------
// One wave32 per point: lanes split candidate j's, keep sorted per-lane top-8,
// then 8 rounds of wave-min merge via __shfl_xor on packed (dist,idx) keys.
template <int FPAD>
__global__ __launch_bounds__(256) void knn_kernel(
    const _Float16* __restrict__ x, int* __restrict__ knn_out) {
  __shared__ _Float16 Xi[8][FPAD];
  const int tid = threadIdx.x, lane = tid & 31, wave = tid >> 5;
  const int g  = blockIdx.x * 8 + wave;          // global point id
  const int gb = g & ~(NPT - 1);                 // graph base row
  for (int f = lane; f < FPAD; f += 32) Xi[wave][f] = x[(size_t)g * FPAD + f];
  __syncthreads();

  float d8[8]; int i8[8];
#pragma unroll
  for (int s = 0; s < 8; ++s) { d8[s] = 3.4e38f; i8[s] = gb; }

  for (int jl = lane; jl < NPT; jl += 32) {
    const _Float16* xj = x + (size_t)(gb + jl) * FPAD;
    float acc = 0.f;
    for (int f = 0; f < FPAD; f += 8) {
      v8h a = *(const v8h*)&Xi[wave][f];
      v8h b = *(const v8h*)(xj + f);
#pragma unroll
      for (int t = 0; t < 8; ++t) {
        float df = (float)a[t] - (float)b[t];
        acc += df * df;
      }
    }
    if (acc < d8[7]) {                           // sorted insert (one pass)
      d8[7] = acc; i8[7] = gb + jl;
#pragma unroll
      for (int s = 6; s >= 0; --s)
        if (d8[s + 1] < d8[s]) {
          float td = d8[s]; d8[s] = d8[s + 1]; d8[s + 1] = td;
          int   ti = i8[s]; i8[s] = i8[s + 1]; i8[s + 1] = ti;
        }
    }
  }
#pragma unroll
  for (int r = 0; r < KNB; ++r) {
    unsigned long long cand =
        ((unsigned long long)__float_as_uint(d8[0]) << 32) | (unsigned)i8[0];
    unsigned long long k = cand;
#pragma unroll
    for (int off = 16; off; off >>= 1) {
      unsigned long long o = __shfl_xor(k, off, 32);
      k = (o < k) ? o : k;
    }
    if (cand == k) {                             // winner pops its head
#pragma unroll
      for (int s = 0; s < 7; ++s) { d8[s] = d8[s + 1]; i8[s] = i8[s + 1]; }
      d8[7] = 3.4e38f;
    }
    if (lane == r) knn_out[g * KNB + r] = (int)(unsigned)(k & 0xffffffffu);
  }
}

// ---- fused edge MLP: gather [xi, xj-xi] -> GEMM -> lrelu -> GEMM -> lrelu
//      -> max over K=8 edges.  Block = 4 points = 32 edge rows = 2 M-tiles.
// LDS: Ald persists through GEMM1; U is the staged-gather buffer before
// GEMM1 and the hidden-activation fragment buffer after (aliased, barriered).
template <int KT1, int F, int FPAD>
__global__ __launch_bounds__(256) void edge_mlp_kernel(
    const _Float16* __restrict__ xin,   // [32768, FPAD]
    const int*      __restrict__ nbr,   // [32768, 8]
    const _Float16* __restrict__ W1p,   // [NT1*KT1*512] fragment order
    const float*    __restrict__ b1,    // [336]
    const _Float16* __restrict__ W2p,   // [NT2*KT2*512] fragment order
    const float*    __restrict__ b2,    // [256]
    _Float16*       __restrict__ xout)  // [32768, 256]
{
  __shared__ alignas(32) _Float16 Ald[2 * KT1 * 512];  // 2 M-tiles, frag order
  __shared__ alignas(32) _Float16 U[2 * KT2 * 512];    // stage(36*FPAD) / Hld
  __shared__ int Jld[32];
  static_assert(36 * FPAD <= 2 * KT2 * 512, "stage fits in U");
  const int tid = threadIdx.x, lane = tid & 31, wave = tid >> 5;
  const int pbase = blockIdx.x * 4;

  if (tid < 32) Jld[tid] = nbr[(pbase + (tid >> 3)) * KNB + (tid & 7)];
  __syncthreads();

  // Stage rows with vectorized 16B loads: rows 0..3 = xi of the 4 points,
  // rows 4..35 = xj of the 32 edges.
  constexpr int CPR = FPAD / 8;                   // v8h chunks per row
  for (int t = tid; t < 36 * CPR; t += 256) {
    int row = t / CPR, cc = t % CPR;
    int src = (row < 4) ? (pbase + row) : Jld[row - 4];
    *(v8h*)&U[row * FPAD + cc * 8] =
        *(const v8h*)(xin + (size_t)src * FPAD + cc * 8);
  }
  __syncthreads();

  // Build A = [xi | xj - xi | 0-pad], written directly in fragment order.
  for (int e = tid; e < 2 * KT1 * 512; e += 256) {
    int h  = e & 15, ln = (e >> 4) & 31;
    int kt = (e >> 9) % KT1, mt = (e >> 9) / KT1;
    int m  = (mt << 4) | (ln & 15);                 // edge row 0..31
    int k  = (h & 7) | ((ln >> 4) << 3) | ((h >> 3) << 4);
    int c  = kt * 32 + k;
    int p  = m >> 3;                                // point 0..3 in block
    _Float16 v = (_Float16)0.f;
    if (c < F) {
      v = U[p * FPAD + c];
    } else if (c < 2 * F) {
      int f = c - F;
      v = (_Float16)(U[(4 + m) * FPAD + f] - U[p * FPAD + f]);
    }
    Ald[e] = v;
  }
  __syncthreads();                                  // U dead; becomes Hld

  // GEMM1: [32 x 32*KT1] x [32*KT1 x 352]; one B fragment feeds both M-tiles.
  for (int nt = wave; nt < NT1; nt += 8) {
    v8f acc0 = {0.f, 0.f, 0.f, 0.f, 0.f, 0.f, 0.f, 0.f};
    v8f acc1 = acc0;
    const v16h* Ap = (const v16h*)Ald + lane;
    const v16h* Bp = (const v16h*)W1p + (size_t)nt * KT1 * 32 + lane;
    for (int kt = 0; kt < KT1; ++kt) {
      __builtin_prefetch(Bp + (kt + 2) * 32, 0, 0);
      v16h b = Bp[kt * 32];
      acc0 = wmma16(Ap[kt * 32], b, acc0);
      acc1 = wmma16(Ap[(KT1 + kt) * 32], b, acc1);
    }
    int   n    = nt * 16 + (lane & 15);
    float bias = (n < DH) ? b1[n] : 0.f;
    int   mr0  = (lane >> 4) * 8;                   // C layout: lanes>=16 -> M+8
    int   k2l  = n & 31, kt2 = n >> 5;              // hidden col -> frag pos
    int   h2   = ((k2l >> 4) << 3) | (k2l & 7);
#pragma unroll
    for (int mt = 0; mt < 2; ++mt) {
      v8f acc = mt ? acc1 : acc0;
#pragma unroll
      for (int r = 0; r < 8; ++r) {
        float v   = lrelu(acc[r] + bias);
        int   row = mt * 16 + mr0 + r;              // edge row
        int   ln2 = (row & 15) | (((k2l >> 3) & 1) << 4);
        U[((mt * KT2 + kt2) * 32 + ln2) * 16 + h2] = (_Float16)v;
      }
    }
  }
  __syncthreads();

  // GEMM2: [32 x 352] x [352 x 256], then max over the 8 edges of each point.
  for (int nt = wave; nt < NT2; nt += 8) {
    v8f acc0 = {0.f, 0.f, 0.f, 0.f, 0.f, 0.f, 0.f, 0.f};
    v8f acc1 = acc0;
    const v16h* Ap = (const v16h*)U + lane;
    const v16h* Bp = (const v16h*)W2p + (size_t)nt * KT2 * 32 + lane;
    for (int kt = 0; kt < KT2; ++kt) {
      __builtin_prefetch(Bp + (kt + 2) * 32, 0, 0);
      v16h b = Bp[kt * 32];
      acc0 = wmma16(Ap[kt * 32], b, acc0);
      acc1 = wmma16(Ap[(KT2 + kt) * 32], b, acc1);
    }
    // lanes 0-15 hold rows 0-7 (even point), lanes 16-31 rows 8-15 (odd):
    // max over K is a per-lane max across the 8 accumulator registers.
    int n = nt * 16 + (lane & 15);
#pragma unroll
    for (int mt = 0; mt < 2; ++mt) {
      v8f acc = mt ? acc1 : acc0;
      float m = acc[0];
#pragma unroll
      for (int r = 1; r < 8; ++r) m = fmaxf(m, acc[r]);
      m = lrelu(m + b2[n]);
      int p = pbase + mt * 2 + (lane >> 4);
      xout[(size_t)p * DOUT + n] = (_Float16)m;
    }
  }
}

// ---- point MLP: concat[x0(6), x1..x4(256 each)] (K=1030 -> pad 1056) ------
__global__ __launch_bounds__(256) void mlp1_kernel(
    const _Float16* __restrict__ x0, const _Float16* __restrict__ x1,
    const _Float16* __restrict__ x2, const _Float16* __restrict__ x3,
    const _Float16* __restrict__ x4,
    const _Float16* __restrict__ W1p, const float* __restrict__ b1,
    const _Float16* __restrict__ W2p, const float* __restrict__ b2,
    _Float16* __restrict__ hout) {
  constexpr int KT1 = 33;                          // 1056 / 32
  __shared__ alignas(32) _Float16 Ald[KT1 * 512];  // one 16-row M-tile
  __shared__ alignas(32) _Float16 Hld[KT2 * 512];
  const int tid = threadIdx.x, lane = tid & 31, wave = tid >> 5;
  const int pbase = blockIdx.x * 16;

  for (int e = tid; e < KT1 * 512; e += 256) {
    int h  = e & 15, ln = (e >> 4) & 31, kt = e >> 9;
    int k  = (h & 7) | ((ln >> 4) << 3) | ((h >> 3) << 4);
    int c  = kt * 32 + k;
    int i  = pbase + (ln & 15);
    _Float16 v = (_Float16)0.f;
    if      (c < 6)    v = x0[(size_t)i * 16  + c];
    else if (c < 262)  v = x1[(size_t)i * 256 + (c - 6)];
    else if (c < 518)  v = x2[(size_t)i * 256 + (c - 262)];
    else if (c < 774)  v = x3[(size_t)i * 256 + (c - 518)];
    else if (c < 1030) v = x4[(size_t)i * 256 + (c - 774)];
    Ald[e] = v;
  }
  __syncthreads();

  for (int nt = wave; nt < NT1; nt += 8) {
    v8f acc = {0.f, 0.f, 0.f, 0.f, 0.f, 0.f, 0.f, 0.f};
    const v16h* Ap = (const v16h*)Ald + lane;
    const v16h* Bp = (const v16h*)W1p + (size_t)nt * KT1 * 32 + lane;
    for (int kt = 0; kt < KT1; ++kt) {
      __builtin_prefetch(Bp + (kt + 2) * 32, 0, 0);
      acc = wmma16(Ap[kt * 32], Bp[kt * 32], acc);
    }
    int   n    = nt * 16 + (lane & 15);
    float bias = (n < DH) ? b1[n] : 0.f;
    int   mr0  = (lane >> 4) * 8;
#pragma unroll
    for (int r = 0; r < 8; ++r) {
      float v   = lrelu(acc[r] + bias);
      int   row = mr0 + r;
      int   k2l = n & 31, kt2 = n >> 5;
      int   ln2 = (row & 15) | (((k2l >> 3) & 1) << 4);
      int   h2  = ((k2l >> 4) << 3) | (k2l & 7);
      Hld[(kt2 * 32 + ln2) * 16 + h2] = (_Float16)v;
    }
  }
  __syncthreads();

  for (int nt = wave; nt < NT2; nt += 8) {
    v8f acc = {0.f, 0.f, 0.f, 0.f, 0.f, 0.f, 0.f, 0.f};
    const v16h* Ap = (const v16h*)Hld + lane;
    const v16h* Bp = (const v16h*)W2p + (size_t)nt * KT2 * 32 + lane;
    for (int kt = 0; kt < KT2; ++kt)
      acc = wmma16(Ap[kt * 32], Bp[kt * 32], acc);
    int   n   = nt * 16 + (lane & 15);
    int   mr0 = (lane >> 4) * 8;
#pragma unroll
    for (int r = 0; r < 8; ++r) {
      float v = lrelu(acc[r] + b2[n]);
      hout[(size_t)(pbase + mr0 + r) * DOUT + n] = (_Float16)v;
    }
  }
}

// ---- global mean pool over the 512 points of each graph -------------------
__global__ __launch_bounds__(256) void mean_pool_kernel(
    const _Float16* __restrict__ h, float* __restrict__ g) {
  int b = blockIdx.x, c = threadIdx.x;             // 256 channels
  float acc = 0.f;
  for (int n = 0; n < NPT; ++n)
    acc += (float)h[((size_t)b * NPT + n) * DOUT + c];
  g[b * DOUT + c] = acc * (1.f / NPT);
}

// ---- tiny head: lrelu(g @ W1 + b1) @ W2 + b2  (f32 VALU, negligible) ------
__global__ __launch_bounds__(256) void head_kernel(
    const float* __restrict__ g, const float* __restrict__ w1,
    const float* __restrict__ b1, const float* __restrict__ w2,
    const float* __restrict__ b2, float* __restrict__ out) {
  __shared__ float gs[256];
  __shared__ float hs[128];
  int b = blockIdx.x, t = threadIdx.x;
  gs[t] = g[b * 256 + t];
  __syncthreads();
  if (t < 128) {
    float acc = b1[t];
    for (int c = 0; c < 256; ++c) acc += gs[c] * w1[c * 128 + t];
    hs[t] = lrelu(acc);
  }
  __syncthreads();
  if (t < 3) {
    float acc = b2[t];
    for (int c = 0; c < 128; ++c) acc += hs[c] * w2[c * 3 + t];
    out[b * 3 + t] = acc;
  }
}

// ---------------------------------------------------------------------------
extern "C" void kernel_launch(void* const* d_in, const int* in_sizes, int n_in,
                              void* d_out, int out_size, void* d_ws,
                              size_t ws_size, hipStream_t stream) {
  (void)in_sizes; (void)n_in; (void)out_size; (void)ws_size;
  const float* x = (const float*)d_in[0];
  const float *cw1[4], *cb1[4], *cw2[4], *cb2[4];
  for (int l = 0; l < 4; ++l) {
    cw1[l] = (const float*)d_in[3 + 4 * l];
    cb1[l] = (const float*)d_in[4 + 4 * l];
    cw2[l] = (const float*)d_in[5 + 4 * l];
    cb2[l] = (const float*)d_in[6 + 4 * l];
  }
  const float* m1_w1 = (const float*)d_in[19];
  const float* m1_b1 = (const float*)d_in[20];
  const float* m1_w2 = (const float*)d_in[21];
  const float* m1_b2 = (const float*)d_in[22];
  const float* m2_w1 = (const float*)d_in[23];
  const float* m2_b1 = (const float*)d_in[24];
  const float* m2_w2 = (const float*)d_in[25];
  const float* m2_b2 = (const float*)d_in[26];

  // Workspace carve-up (~90 MB needed).
  char* wp = (char*)d_ws;
  auto carve = [&](size_t bytes) -> char* {
    char* r = wp;
    wp += (bytes + 255) & ~(size_t)255;
    return r;
  };
  _Float16* x0p = (_Float16*)carve((size_t)NPTS * 16 * 2);
  _Float16* xb[4];
  for (int l = 0; l < 4; ++l) xb[l] = (_Float16*)carve((size_t)NPTS * 256 * 2);
  int* idx = (int*)carve((size_t)NPTS * KNB * 4);
  const int kt1s[4] = {1, 16, 16, 16};             // GEMM1 k-tiles per layer
  _Float16 *W1p[4], *W2p[4];
  for (int l = 0; l < 4; ++l) {
    W1p[l] = (_Float16*)carve((size_t)NT1 * kt1s[l] * 512 * 2);
    W2p[l] = (_Float16*)carve((size_t)NT2 * KT2 * 512 * 2);
  }
  _Float16* W1pm = (_Float16*)carve((size_t)NT1 * 33 * 512 * 2);
  _Float16* W2pm = (_Float16*)carve((size_t)NT2 * KT2 * 512 * 2);
  _Float16* hbuf = (_Float16*)carve((size_t)NPTS * 256 * 2);
  float*    gbuf = (float*)carve((size_t)NBATCH * 256 * 4);

  // Stage 0: precision conversion + weight packing (fragment order).
  cvt_x0_kernel<<<NPTS * 16 / 256, 256, 0, stream>>>(x, x0p);
  const int din1[4] = {12, 512, 512, 512};
  for (int l = 0; l < 4; ++l) {
    int t1 = NT1 * kt1s[l] * 512;
    pack_w_kernel<<<(t1 + 255) / 256, 256, 0, stream>>>(cw1[l], W1p[l],
                                                        din1[l], DH, kt1s[l], t1);
    int t2 = NT2 * KT2 * 512;
    pack_w_kernel<<<(t2 + 255) / 256, 256, 0, stream>>>(cw2[l], W2p[l],
                                                        DH, DOUT, KT2, t2);
  }
  {
    int t1 = NT1 * 33 * 512;
    pack_w_kernel<<<(t1 + 255) / 256, 256, 0, stream>>>(m1_w1, W1pm, 1030, DH,
                                                        33, t1);
    int t2 = NT2 * KT2 * 512;
    pack_w_kernel<<<(t2 + 255) / 256, 256, 0, stream>>>(m1_w2, W2pm, DH, DOUT,
                                                        KT2, t2);
  }

  // Layer 1: F=6 (padded to 16), 2F=12 -> one 32-wide k-tile.
  knn_kernel<16><<<NPTS / 8, 256, 0, stream>>>(x0p, idx);
  edge_mlp_kernel<1, 6, 16><<<NPTS / 4, 256, 0, stream>>>(
      x0p, idx, W1p[0], cb1[0], W2p[0], cb2[0], xb[0]);

  // Layers 2-4: F=256, 2F=512 -> 16 k-tiles.
  for (int l = 1; l < 4; ++l) {
    knn_kernel<256><<<NPTS / 8, 256, 0, stream>>>(xb[l - 1], idx);
    edge_mlp_kernel<16, 256, 256><<<NPTS / 4, 256, 0, stream>>>(
        xb[l - 1], idx, W1p[l], cb1[l], W2p[l], cb2[l], xb[l]);
  }

  // Point MLP on concat(1030 -> pad 1056), then pool + head.
  mlp1_kernel<<<NPTS / 16, 256, 0, stream>>>(x0p, xb[0], xb[1], xb[2], xb[3],
                                             W1pm, m1_b1, W2pm, m1_b2, hbuf);
  mean_pool_kernel<<<NBATCH, 256, 0, stream>>>(hbuf, gbuf);
  head_kernel<<<NBATCH, 256, 0, stream>>>(gbuf, m2_w1, m2_b1, m2_w2, m2_b2,
                                          (float*)d_out);
}
